// SpatialBlock_42889543418186
// MI455X (gfx1250) — matrix operand order repaired and checked by reading, not compiled, not run on hardware
//
#include <hip/hip_runtime.h>

// ---------------- problem constants ----------------
#define NB   8          // batch
#define TT   12         // time
#define NN   512        // nodes
#define FF   288        // features (K dim)
#define HH   3          // heads
#define DD   32         // head dim
#define EE   8192       // edges per graph
#define MM   96         // T*B
#define NM   (NN*MM)    // 49152 GEMM rows
#define NMH  (NN*FF)    // per-graph el/er size: 512*288 (m*3+h flattened = 288)
#define OUT1 (NB*TT*NN*FF)  // 14,155,776 floats (first output)

// ---------------- WMMA types / helpers ----------------
typedef __bf16 bf16;
typedef bf16  bf16x8  __attribute__((ext_vector_type(8)));
typedef bf16  bf16x16 __attribute__((ext_vector_type(16)));
typedef float f32x8   __attribute__((ext_vector_type(8)));

__device__ __forceinline__ unsigned short f2bf_rne(float x) {
    unsigned u = __float_as_uint(x);
    u += 0x7FFFu + ((u >> 16) & 1u);
    return (unsigned short)(u >> 16);
}
__device__ __forceinline__ float bf2f(unsigned short h) {
    return __uint_as_float(((unsigned)h) << 16);
}
__device__ __forceinline__ void split_store(float v, unsigned short* hi, unsigned short* lo, size_t i) {
    unsigned short h = f2bf_rne(v);
    hi[i] = h;
    lo[i] = f2bf_rne(v - bf2f(h));
}

// A-fragment (16x32 bf16, MxK): lane l=L&15 holds row l, half=L>>4 selects K phase.
// VGPR0..3 = K[k0+half*8 .. +7], VGPR4..7 = K[k0+16+half*8 .. +7]  (contiguous 16B chunks)
__device__ __forceinline__ bf16x16 frag_a(const unsigned short* __restrict__ row, int k0, int half) {
    bf16x8 a = *reinterpret_cast<const bf16x8*>(row + k0 + half * 8);
    bf16x8 b = *reinterpret_cast<const bf16x8*>(row + k0 + 16 + half * 8);
    return __builtin_shufflevector(a, b, 0,1,2,3,4,5,6,7,8,9,10,11,12,13,14,15);
}
// B-fragment (32x16 bf16, KxN): lane l holds column l; K = k0+half*8 .. +7, k0+16+half*8 .. +7
// (row = column-major source => we stage W transposed as (col, k))
__device__ __forceinline__ bf16x16 frag_b(const unsigned short* __restrict__ row, int k0, int half) {
    bf16x8 a = *reinterpret_cast<const bf16x8*>(row + k0 + half * 8);
    bf16x8 b = *reinterpret_cast<const bf16x8*>(row + k0 + 16 + half * 8);
    return __builtin_shufflevector(a, b, 0,1,2,3,4,5,6,7,8,9,10,11,12,13,14,15);
}
__device__ __forceinline__ f32x8 wmma_bf(bf16x16 a, bf16x16 b, f32x8 c) {
    return __builtin_amdgcn_wmma_f32_16x16x32_bf16(false, a, false, b, (short)0, c, false, false);
}

// ---------------- prep kernels ----------------
// A[(n*96 + t*8 + b), f] = x_in[b,t,n,f] + cov[b,t,n,f]; split to bf16 hi/lo
__global__ void prep_A(const float* __restrict__ x_in, const float* __restrict__ cov,
                       unsigned short* __restrict__ Ahi, unsigned short* __restrict__ Alo) {
    int i = blockIdx.x * 256 + threadIdx.x;
    if (i >= NM * FF) return;
    int f = i % FF;
    int j = i / FF;          // row 0..49151
    int n = j / MM;
    int m = j % MM;
    int t = m >> 3;          // m = t*8 + b
    int b = m & 7;
    size_t s = ((size_t)(b * TT + t) * NN + n) * FF + f;
    split_store(x_in[s] + cov[s], Ahi, Alo, (size_t)i);
}

// Wt[c][k] = W_{c/96}[k][c%96], c in 0..287 ; split hi/lo
__global__ void prep_W(const float* __restrict__ Wd, const float* __restrict__ Wm,
                       const float* __restrict__ Ws,
                       unsigned short* __restrict__ Whi, unsigned short* __restrict__ Wlo) {
    int i = blockIdx.x * 256 + threadIdx.x;
    if (i >= FF * FF) return;
    int k = i % FF;
    int c = i / FF;
    int g = c / MM;
    int cl = c - g * MM;
    const float* W = (g == 0) ? Wd : (g == 1) ? Wm : Ws;
    split_store(W[(size_t)k * MM + cl], Whi, Wlo, (size_t)i);
}

// cov0[b,n,f] = cov[b,0,n,f]; split hi/lo
__global__ void prep_cov0(const float* __restrict__ cov,
                          unsigned short* __restrict__ Chi, unsigned short* __restrict__ Clo) {
    int i = blockIdx.x * 256 + threadIdx.x;
    if (i >= NB * NN * FF) return;
    int b = i / (NN * FF);
    int rem = i % (NN * FF);
    size_t s = (size_t)b * TT * NN * FF + rem;
    split_store(cov[s], Chi, Clo, (size_t)i);
}

// ---------------- feature GEMM: H = A(49152x288) @ Wt^T, bf16x3 split ----------------
// wave computes 16 rows x 96 cols (one graph). waves = 3072*3 = 9216 -> 1152 blocks.
__global__ __launch_bounds__(256) void gemm_h(const unsigned short* __restrict__ Ahi,
                                              const unsigned short* __restrict__ Alo,
                                              const unsigned short* __restrict__ Whi,
                                              const unsigned short* __restrict__ Wlo,
                                              float* __restrict__ Hout) {
    int wave = blockIdx.x * 8 + (threadIdx.x >> 5);
    int lane = threadIdx.x & 31;
    int l = lane & 15, half = lane >> 4;
    int rt = wave / 3;       // row tile 0..3071
    int g  = wave % 3;       // graph
    const unsigned short* arh = Ahi + (size_t)(rt * 16 + l) * FF;
    const unsigned short* arl = Alo + (size_t)(rt * 16 + l) * FF;

    f32x8 acc[6];
    f32x8 z = {0.f, 0.f, 0.f, 0.f, 0.f, 0.f, 0.f, 0.f};
#pragma unroll
    for (int t = 0; t < 6; ++t) acc[t] = z;

    for (int k0 = 0; k0 < FF; k0 += 32) {
        bf16x16 ah = frag_a(arh, k0, half);
        bf16x16 al = frag_a(arl, k0, half);
#pragma unroll
        for (int t = 0; t < 6; ++t) {
            size_t wrow = (size_t)(g * MM + t * 16 + l) * FF;
            bf16x16 bh = frag_b(Whi + wrow, k0, half);
            bf16x16 bl = frag_b(Wlo + wrow, k0, half);
            acc[t] = wmma_bf(ah, bh, acc[t]);
            acc[t] = wmma_bf(ah, bl, acc[t]);
            acc[t] = wmma_bf(al, bh, acc[t]);
        }
    }
    int rbase = rt * 16 + 8 * half;
#pragma unroll
    for (int t = 0; t < 6; ++t)
#pragma unroll
        for (int r = 0; r < 8; ++r)
            Hout[(size_t)(rbase + r) * FF + g * MM + t * 16 + l] = acc[t][r];
}

// ---------------- el/er: per (g,n,m,h) 32-dot with attn vectors ----------------
__global__ void elr_kernel(const float* __restrict__ Hout,
                           const float* __restrict__ ald, const float* __restrict__ ard,
                           const float* __restrict__ alm, const float* __restrict__ arm,
                           const float* __restrict__ als, const float* __restrict__ ars,
                           float* __restrict__ el, float* __restrict__ er) {
    int i = blockIdx.x * 256 + threadIdx.x;
    if (i >= 3 * NMH) return;
    int g = i / NMH;
    int rem = i % NMH;
    int n = rem / FF;
    int q = rem % FF;        // q = m*3 + h
    int m = q / 3, h = q % 3;
    const float* al = (g == 0) ? ald : (g == 1) ? alm : als;
    const float* ar = (g == 0) ? ard : (g == 1) ? arm : ars;
    const float* hp = Hout + (size_t)(n * MM + m) * FF + g * MM + h * DD;
    float sl = 0.f, sr = 0.f;
#pragma unroll 8
    for (int d = 0; d < DD; ++d) {
        float v = hp[d];
        sl += v * al[h * DD + d];
        sr += v * ar[h * DD + d];
    }
    el[i] = sl;
    er[i] = sr;
}

// ---------------- deterministic CSR by dst ----------------
__global__ void edge_count(const int* __restrict__ dd, const int* __restrict__ dm,
                           const int* __restrict__ ds, int* __restrict__ deg) {
    int e = blockIdx.x * 256 + threadIdx.x;
    int g = blockIdx.y;
    if (e >= EE) return;
    const int* dst = (g == 0) ? dd : (g == 1) ? dm : ds;
    atomicAdd(&deg[g * NN + dst[e]], 1);
}
__global__ void edge_scan(const int* __restrict__ deg, int* __restrict__ off) {
    int g = blockIdx.x;
    if (threadIdx.x != 0) return;
    int acc = 0;
    off[g * (NN + 1)] = 0;
    for (int n = 0; n < NN; ++n) {
        acc += deg[g * NN + n];
        off[g * (NN + 1) + n + 1] = acc;
    }
}
// stable rank: rank(e) = #{j<e : dst[j]==dst[e]}  (deterministic, no atomics)
__global__ void edge_place(const int* __restrict__ dd, const int* __restrict__ dm,
                           const int* __restrict__ ds, const int* __restrict__ off,
                           int* __restrict__ elist) {
    int e = blockIdx.x * 256 + threadIdx.x;
    int g = blockIdx.y;
    if (e >= EE) return;
    const int* dst = (g == 0) ? dd : (g == 1) ? dm : ds;
    int d = dst[e];
    int rank = 0;
    for (int j = 0; j < e; ++j) rank += (dst[j] == d);
    elist[g * EE + off[g * (NN + 1) + d] + rank] = e;
}

// ---------------- segment max + sum(exp) per (g, n, m, h) — gather, no atomics ----------------
__global__ __launch_bounds__(288) void maxsum_kernel(
        const int* __restrict__ sd, const int* __restrict__ sm_, const int* __restrict__ ss,
        const int* __restrict__ off, const int* __restrict__ elist,
        const float* __restrict__ el, const float* __restrict__ er,
        float* __restrict__ maxb, float* __restrict__ sumb) {
    int n = blockIdx.x, g = blockIdx.y;
    int t = threadIdx.x;             // t = m*3 + h, 0..287
    const int* src = (g == 0) ? sd : (g == 1) ? sm_ : ss;
    int o0 = off[g * (NN + 1) + n];
    int dg = off[g * (NN + 1) + n + 1] - o0;
    float erv = er[(size_t)g * NMH + n * FF + t];
    float mx = -INFINITY;
    for (int i = 0; i < dg; ++i) {
        int s = src[elist[g * EE + o0 + i]];
        float ev = el[(size_t)g * NMH + s * FF + t] + erv;
        ev = ev > 0.f ? ev : 0.2f * ev;
        mx = fmaxf(mx, ev);
    }
    float sum = 0.f;
    for (int i = 0; i < dg; ++i) {
        int s = src[elist[g * EE + o0 + i]];
        float ev = el[(size_t)g * NMH + s * FF + t] + erv;
        ev = ev > 0.f ? ev : 0.2f * ev;
        sum += __expf(ev - mx);
    }
    maxb[(size_t)g * NMH + n * FF + t] = mx;
    sumb[(size_t)g * NMH + n * FF + t] = sum;
}

// ---------------- alpha-weighted aggregation + residual + bias epilogue ----------------
// block = (n, m), 96 threads = output channels c = h*32+d of graph g
__global__ __launch_bounds__(96) void agg_kernel(
        const int* __restrict__ sd, const int* __restrict__ sm_, const int* __restrict__ ss,
        const int* __restrict__ off, const int* __restrict__ elist,
        const float* __restrict__ el, const float* __restrict__ er,
        const float* __restrict__ maxb, const float* __restrict__ sumb,
        const float* __restrict__ Hout, const float* __restrict__ x_in,
        const float* __restrict__ bd, const float* __restrict__ bm, const float* __restrict__ bs,
        float* __restrict__ out) {
    int c = threadIdx.x;             // 0..95
    int h = c >> 5;
    int n = blockIdx.x / MM;
    int m = blockIdx.x % MM;
    int g = blockIdx.y;
    const int* src = (g == 0) ? sd : (g == 1) ? sm_ : ss;
    const float* bias = (g == 0) ? bd : (g == 1) ? bm : bs;
    int o0 = off[g * (NN + 1) + n];
    int dg = off[g * (NN + 1) + n + 1] - o0;
    size_t inh = (size_t)g * NMH + n * FF + m * 3 + h;
    float mx = maxb[inh];
    float inv = (dg > 0) ? 1.f / sumb[inh] : 0.f;
    float erv = er[inh];
    float acc = 0.f;
    for (int i = 0; i < dg; ++i) {
        int s = src[elist[g * EE + o0 + i]];
        float ev = el[(size_t)g * NMH + s * FF + m * 3 + h] + erv;
        ev = ev > 0.f ? ev : 0.2f * ev;
        float alpha = __expf(ev - mx) * inv;
        acc += alpha * Hout[(size_t)(s * MM + m) * FF + g * MM + c];
    }
    int b = m & 7, t = m >> 3;
    size_t oidx = ((size_t)(b * TT + t) * NN + n) * FF + g * MM + c;
    out[oidx] = x_in[oidx] + bias[c] + acc;
}

// ---------------- Gram matrix S[b] = cov0[b] @ cov0[b]^T (bf16x3 WMMA) ----------------
// wave computes 16 rows x 64 cols. waves = 8*32*8 = 2048 -> 256 blocks.
__global__ __launch_bounds__(256) void gram_kernel(const unsigned short* __restrict__ Chi,
                                                   const unsigned short* __restrict__ Clo,
                                                   float* __restrict__ S) {
    int wave = blockIdx.x * 8 + (threadIdx.x >> 5);
    int lane = threadIdx.x & 31;
    int l = lane & 15, half = lane >> 4;
    int b = wave >> 8;
    int rem = wave & 255;
    int rt = rem >> 3, cg = rem & 7;

    const unsigned short* arh = Chi + (size_t)(b * NN + rt * 16 + l) * FF;
    const unsigned short* arl = Clo + (size_t)(b * NN + rt * 16 + l) * FF;

    f32x8 acc[4];
    f32x8 z = {0.f, 0.f, 0.f, 0.f, 0.f, 0.f, 0.f, 0.f};
#pragma unroll
    for (int t = 0; t < 4; ++t) acc[t] = z;

    for (int k0 = 0; k0 < FF; k0 += 32) {
        bf16x16 ah = frag_a(arh, k0, half);
        bf16x16 al = frag_a(arl, k0, half);
#pragma unroll
        for (int t = 0; t < 4; ++t) {
            size_t crow = (size_t)(b * NN + cg * 64 + t * 16 + l) * FF;  // B^T row = cov0 row
            bf16x16 bh = frag_b(Chi + crow, k0, half);
            bf16x16 bl = frag_b(Clo + crow, k0, half);
            acc[t] = wmma_bf(ah, bh, acc[t]);
            acc[t] = wmma_bf(ah, bl, acc[t]);
            acc[t] = wmma_bf(al, bh, acc[t]);
        }
    }
    int rbase = rt * 16 + 8 * half;
#pragma unroll
    for (int t = 0; t < 4; ++t)
#pragma unroll
        for (int r = 0; r < 8; ++r)
            S[((size_t)b * NN + rbase + r) * NN + cg * 64 + t * 16 + l] = acc[t][r];
}

// ---------------- row softmax over 512 cols, write 12 tiled copies ----------------
__global__ __launch_bounds__(256) void softmax_tile(const float* __restrict__ S,
                                                    float* __restrict__ attn_out) {
    __shared__ float red[256];
    int r = blockIdx.x, b = blockIdx.y, t = threadIdx.x;
    const float* row = S + ((size_t)b * NN + r) * NN;
    float v0 = row[t], v1 = row[t + 256];
    float mx = fmaxf(v0, v1);
    red[t] = mx;
    __syncthreads();
    for (int s = 128; s > 0; s >>= 1) {
        if (t < s) red[t] = fmaxf(red[t], red[t + s]);
        __syncthreads();
    }
    mx = red[0];
    __syncthreads();
    float e0 = __expf(v0 - mx), e1 = __expf(v1 - mx);
    red[t] = e0 + e1;
    __syncthreads();
    for (int s = 128; s > 0; s >>= 1) {
        if (t < s) red[t] += red[t + s];
        __syncthreads();
    }
    float inv = 1.f / red[0];
    float a0 = e0 * inv, a1 = e1 * inv;
    for (int rep = 0; rep < TT; ++rep) {
        float* o = attn_out + ((size_t)(rep * NB + b) * NN + r) * NN;
        o[t] = a0;
        o[t + 256] = a1;
    }
}

// ---------------- launch ----------------
extern "C" void kernel_launch(void* const* d_in, const int* in_sizes, int n_in,
                              void* d_out, int out_size, void* d_ws, size_t ws_size,
                              hipStream_t stream) {
    const float* x_in = (const float*)d_in[0];
    const float* cov  = (const float*)d_in[1];
    const int *src_d = (const int*)d_in[2],  *dst_d = (const int*)d_in[3];
    const float *W_d = (const float*)d_in[4], *al_d = (const float*)d_in[5],
                *ar_d = (const float*)d_in[6], *b_d = (const float*)d_in[7];
    const int *src_m = (const int*)d_in[8],  *dst_m = (const int*)d_in[9];
    const float *W_m = (const float*)d_in[10], *al_m = (const float*)d_in[11],
                *ar_m = (const float*)d_in[12], *b_m = (const float*)d_in[13];
    const int *src_s = (const int*)d_in[14], *dst_s = (const int*)d_in[15];
    const float *W_s = (const float*)d_in[16], *al_s = (const float*)d_in[17],
                *ar_s = (const float*)d_in[18], *b_s = (const float*)d_in[19];

    // workspace carve-out (~132 MB)
    char* ws = (char*)d_ws;
    size_t o = 0;
    auto carve = [&](size_t bytes) { size_t p = o; o = (o + bytes + 255) & ~(size_t)255; return p; };
    unsigned short* Ahi = (unsigned short*)(ws + carve((size_t)NM * FF * 2));
    unsigned short* Alo = (unsigned short*)(ws + carve((size_t)NM * FF * 2));
    unsigned short* Whi = (unsigned short*)(ws + carve((size_t)FF * FF * 2));
    unsigned short* Wlo = (unsigned short*)(ws + carve((size_t)FF * FF * 2));
    float* Hout = (float*)(ws + carve((size_t)NM * FF * 4));
    float* el   = (float*)(ws + carve((size_t)3 * NMH * 4));
    float* er   = (float*)(ws + carve((size_t)3 * NMH * 4));
    float* maxb = (float*)(ws + carve((size_t)3 * NMH * 4));
    float* sumb = (float*)(ws + carve((size_t)3 * NMH * 4));
    int* deg    = (int*)(ws + carve((size_t)3 * NN * 4));
    int* off    = (int*)(ws + carve((size_t)3 * (NN + 1) * 4));
    int* elist  = (int*)(ws + carve((size_t)3 * EE * 4));
    unsigned short* Chi = (unsigned short*)(ws + carve((size_t)NB * NN * FF * 2));
    unsigned short* Clo = (unsigned short*)(ws + carve((size_t)NB * NN * FF * 2));
    float* S = (float*)(ws + carve((size_t)NB * NN * NN * 4));

    float* out = (float*)d_out;
    float* attn_out = out + OUT1;

    // --- prep ---
    prep_A<<<(NM * FF + 255) / 256, 256, 0, stream>>>(x_in, cov, Ahi, Alo);
    prep_W<<<(FF * FF + 255) / 256, 256, 0, stream>>>(W_d, W_m, W_s, Whi, Wlo);
    prep_cov0<<<(NB * NN * FF + 255) / 256, 256, 0, stream>>>(cov, Chi, Clo);

    // --- feature GEMM (WMMA) ---
    gemm_h<<<1152, 256, 0, stream>>>(Ahi, Alo, Whi, Wlo, Hout);

    // --- el / er ---
    elr_kernel<<<(3 * NMH + 255) / 256, 256, 0, stream>>>(Hout, al_d, ar_d, al_m, ar_m, al_s, ar_s, el, er);

    // --- CSR build (deterministic) ---
    hipMemsetAsync(deg, 0, (size_t)3 * NN * 4, stream);
    edge_count<<<dim3((EE + 255) / 256, 3), 256, 0, stream>>>(dst_d, dst_m, dst_s, deg);
    edge_scan<<<3, 32, 0, stream>>>(deg, off);
    edge_place<<<dim3((EE + 255) / 256, 3), 256, 0, stream>>>(dst_d, dst_m, dst_s, off, elist);

    // --- segment max + sum ---
    maxsum_kernel<<<dim3(NN, 3), 288, 0, stream>>>(src_d, src_m, src_s, off, elist, el, er, maxb, sumb);

    // --- aggregation + residual + bias -> d_out[0:OUT1) ---
    agg_kernel<<<dim3(NN * MM, 3), 96, 0, stream>>>(src_d, src_m, src_s, off, elist, el, er,
                                                    maxb, sumb, Hout, x_in, b_d, b_m, b_s, out);

    // --- attn: Gram (WMMA) + softmax + tile -> d_out[OUT1:) ---
    gram_kernel<<<256, 256, 0, stream>>>(Chi, Clo, S);
    softmax_tile<<<dim3(NN, NB), 256, 0, stream>>>(S, attn_out);
}
